// InvariantPointAttention_4526895530282
// MI455X (gfx1250) — matrix operand order repaired and checked by reading, not compiled, not run on hardware
//
#include <hip/hip_runtime.h>
#include <hip/hip_bf16.h>
#include <cstdint>
#include <cstddef>

#define DEVI __device__ __forceinline__

typedef __attribute__((ext_vector_type(16))) _Float16 v16h;
typedef __attribute__((ext_vector_type(8)))  _Float16 v8h;
typedef __attribute__((ext_vector_type(8)))  float    v8f;

namespace {
constexpr int kB   = 2;
constexpr int kN   = 1024;
constexpr int kDIM = 768;
constexpr int kH   = 8;
constexpr int kDH  = 64;
constexpr int kPD  = 4;
constexpr int kM   = kB * kN;              // 2048 GEMM rows
constexpr int kSDim = kH * kDH;            // 512
constexpr int kPDim = kH * kPD * 3;        // 96
constexpr int kCat  = kH * (kDH + kPD * 4);// 640
constexpr float kEPS = 1e-8f;
}

// ---------------------------------------------------------------------------
// WMMA fragment helpers (wave32 layouts per CDNA5 ISA 7.12.2)
// A 16x32 f16: lane m=l&15; lanes<16 cover K {0..7,16..23}, lanes>=16 {8..15,24..31}
// -> two contiguous 8-half (16B) loads per lane.
// B 32x16 f16: lane holds column n=l&15; same K order as A: lanes<16 K {0..7,16..23},
// lanes>=16 {8..15,24..31} -> two contiguous 8-half (16B) loads per lane.
// C/D f32: VGPR r -> row r + 8*(l>>4), col l&15.
// ---------------------------------------------------------------------------
DEVI v16h load_a_frag(const _Float16* base, int ldk, int k0, int lane) {
  const int m   = lane & 15;
  const int grp = (lane >> 4) << 3;          // 0 or 8
  const _Float16* p = base + (size_t)m * ldk + k0 + grp;
  v8h lo = *reinterpret_cast<const v8h*>(p);
  v8h hi = *reinterpret_cast<const v8h*>(p + 16);
  v16h r;
#pragma unroll
  for (int i = 0; i < 8; ++i) { r[i] = lo[i]; r[8 + i] = hi[i]; }
  return r;
}

DEVI v16h load_b_frag(const _Float16* base, int ldk, int col0, int k0, int lane) {
  const int n   = lane & 15;
  const int grp = (lane >> 4) << 3;          // 0 or 8
  const _Float16* p = base + (size_t)(col0 + n) * ldk + k0 + grp;
  v8h lo = *reinterpret_cast<const v8h*>(p);
  v8h hi = *reinterpret_cast<const v8h*>(p + 16);
  v16h r;
#pragma unroll
  for (int i = 0; i < 8; ++i) { r[i] = lo[i]; r[8 + i] = hi[i]; }
  return r;
}

DEVI v8f wmma16(v16h a, v16h b, v8f c) {
  return __builtin_amdgcn_wmma_f32_16x16x32_f16(false, a, false, b, (short)0, c,
                                                false, false);
}

DEVI float bperm_xor(float v, int m, int lane) {
  int idx = (lane ^ m) << 2;
  return __builtin_bit_cast(
      float, __builtin_amdgcn_ds_bpermute(idx, __builtin_bit_cast(int, v)));
}
DEVI float redmax16(float v, int lane) {
#pragma unroll
  for (int m = 1; m < 16; m <<= 1) v = fmaxf(v, bperm_xor(v, m, lane));
  return v;
}
DEVI float redsum16(float v, int lane) {
#pragma unroll
  for (int m = 1; m < 16; m <<= 1) v += bperm_xor(v, m, lane);
  return v;
}

// ---------------------------------------------------------------------------
// Kernel: fp32 -> fp16 conversion (grid-stride)
// ---------------------------------------------------------------------------
__global__ void f32_to_f16(const float* __restrict__ in, _Float16* __restrict__ out,
                           int n) {
  for (int i = blockIdx.x * blockDim.x + threadIdx.x; i < n;
       i += gridDim.x * blockDim.x)
    out[i] = (_Float16)in[i];
}

// ---------------------------------------------------------------------------
// Kernel: pwc[h] = 0.5 * softplus(point_weights[h]) * point_scale
// ---------------------------------------------------------------------------
__global__ void softplus_pw(const float* __restrict__ pwin, float* __restrict__ pwc) {
  int h = threadIdx.x;
  if (h < kH) {
    float x  = pwin[h];
    float sp = (x > 20.f) ? x : log1pf(__expf(x));
    const float pscale = rsqrtf(2.f * kPD * 4.5f);   // (2*pd*9/2)^-0.5
    pwc[h] = 0.5f * sp * pscale;
  }
}

// ---------------------------------------------------------------------------
// Kernel: register-blocked f16 GEMM  C(MxNout) = A(MxK) * B(NoutxK)^T (+bias)
// One wave computes (16*MT) x (16*NT) output: MT*NT accumulators, double-
// buffered K-loop (step 64) so next fragments' global_load_b128 issue before
// the current WMMAs -> s_wait_loadcnt overlaps VMEM with the matrix pipe.
// Requires K % 64 == 0.
// ---------------------------------------------------------------------------
template <int MT, int NT>
__global__ void __launch_bounds__(256)
gemm_f16f32(const _Float16* __restrict__ A, const _Float16* __restrict__ Bw,
            const float* __restrict__ bias, float* __restrict__ C, int K,
            int Nout) {
  const int lane  = threadIdx.x & 31;
  const int wid   = (blockIdx.x * blockDim.x + threadIdx.x) >> 5;
  const int tng   = Nout / (16 * NT);
  const int tileN = (wid % tng) * (16 * NT);
  const int tileM = (wid / tng) * (16 * MT);
  const _Float16* arow = A + (size_t)tileM * K;

  const v8f vzero = {};
  v8f acc[MT][NT];
#pragma unroll
  for (int i = 0; i < MT; ++i)
#pragma unroll
    for (int j = 0; j < NT; ++j) acc[i][j] = vzero;

  v16h a0[MT], b0[NT], a1[MT], b1[NT];
#pragma unroll
  for (int i = 0; i < MT; ++i)
    a0[i] = load_a_frag(arow + (size_t)i * 16 * K, K, 0, lane);
#pragma unroll
  for (int j = 0; j < NT; ++j)
    b0[j] = load_b_frag(Bw, K, tileN + j * 16, 0, lane);

  for (int k0 = 0; k0 < K; k0 += 64) {
    if (k0 + 128 < K) {   // stream 2 K-tiles ahead through L2
      __builtin_prefetch(arow + (size_t)(lane & 15) * K + k0 + 128, 0, 3);
      __builtin_prefetch(Bw + (size_t)(tileN + (lane & 15)) * K + k0 + 128, 0, 3);
    }
    // issue loads for second half before consuming first half
#pragma unroll
    for (int i = 0; i < MT; ++i)
      a1[i] = load_a_frag(arow + (size_t)i * 16 * K, K, k0 + 32, lane);
#pragma unroll
    for (int j = 0; j < NT; ++j)
      b1[j] = load_b_frag(Bw, K, tileN + j * 16, k0 + 32, lane);
#pragma unroll
    for (int i = 0; i < MT; ++i)
#pragma unroll
      for (int j = 0; j < NT; ++j) acc[i][j] = wmma16(a0[i], b0[j], acc[i][j]);
    if (k0 + 64 < K) {
#pragma unroll
      for (int i = 0; i < MT; ++i)
        a0[i] = load_a_frag(arow + (size_t)i * 16 * K, K, k0 + 64, lane);
#pragma unroll
      for (int j = 0; j < NT; ++j)
        b0[j] = load_b_frag(Bw, K, tileN + j * 16, k0 + 64, lane);
    }
#pragma unroll
    for (int i = 0; i < MT; ++i)
#pragma unroll
      for (int j = 0; j < NT; ++j) acc[i][j] = wmma16(a1[i], b1[j], acc[i][j]);
  }

  const int rb = ((lane >> 4) << 3);
  const int cl = lane & 15;
#pragma unroll
  for (int i = 0; i < MT; ++i) {
#pragma unroll
    for (int j = 0; j < NT; ++j) {
      const int   col = tileN + j * 16 + cl;
      const float bv  = bias ? bias[col] : 0.f;
#pragma unroll
      for (int r = 0; r < 8; ++r)
        C[(size_t)(tileM + i * 16 + rb + r) * Nout + col] = acc[i][j][r] + bv;
    }
  }
}

// ---------------------------------------------------------------------------
// Kernel: reshape projections, apply rotation+translation to points,
// emit f16 tensors in WMMA-friendly layouts + q2/k2 row norms.
//   qs_h/ks_h : (b,h,n,64)      vsT : (b,h,64,N)
//   qp_h/kp_h : (b,h,n,32) pad  vpT : (b,h,16,N) pad rows 12..15 = 0
// ---------------------------------------------------------------------------
__global__ void reshape_rotate(
    const float* __restrict__ Psq, const float* __restrict__ Psk,
    const float* __restrict__ Psv, const float* __restrict__ Ppq,
    const float* __restrict__ Ppk, const float* __restrict__ Ppv,
    const float* __restrict__ rot, const float* __restrict__ trans,
    _Float16* __restrict__ qs_h, _Float16* __restrict__ ks_h,
    _Float16* __restrict__ vsT, _Float16* __restrict__ qp_h,
    _Float16* __restrict__ kp_h, _Float16* __restrict__ vpT,
    float* __restrict__ q2, float* __restrict__ k2) {
  int idx = blockIdx.x * blockDim.x + threadIdx.x;
  if (idx >= kB * kN * kH) return;
  int h = idx % kH;
  int m = idx / kH;                 // m = b*N + n
  int b = m / kN, n = m % kN;
  int bh = b * kH + h;

  size_t srow = (size_t)m * kSDim + h * kDH;
  size_t o64  = ((size_t)bh * kN + n) * kDH;
#pragma unroll 4
  for (int d = 0; d < kDH; ++d) {
    qs_h[o64 + d] = (_Float16)Psq[srow + d];
    ks_h[o64 + d] = (_Float16)Psk[srow + d];
    vsT[((size_t)bh * kDH + d) * kN + n] = (_Float16)Psv[srow + d];
  }

  const float* R = rot + (size_t)m * 9;
  const float* t = trans + (size_t)m * 3;
  size_t prow = (size_t)m * kPDim + h * kPD * 3;
  size_t o32  = ((size_t)bh * kN + n) * 32;
  float q2s = 0.f, k2s = 0.f;
  float vpv[12];
#pragma unroll
  for (int d = 0; d < kPD; ++d) {
    float pq[3], pk[3], pv[3];
#pragma unroll
    for (int r = 0; r < 3; ++r) {
      pq[r] = Ppq[prow + d * 3 + r];
      pk[r] = Ppk[prow + d * 3 + r];
      pv[r] = Ppv[prow + d * 3 + r];
    }
#pragma unroll
    for (int c = 0; c < 3; ++c) {
      float vq = t[c], vk = t[c], vv = t[c];
#pragma unroll
      for (int r = 0; r < 3; ++r) {
        float rc = R[c * 3 + r];            // rotations[b,n,c,r]
        vq += rc * pq[r];
        vk += rc * pk[r];
        vv += rc * pv[r];
      }
      qp_h[o32 + d * 3 + c] = (_Float16)vq;
      kp_h[o32 + d * 3 + c] = (_Float16)vk;
      vpv[d * 3 + c]        = vv;
      q2s += vq * vq;
      k2s += vk * vk;
    }
  }
#pragma unroll
  for (int e = 12; e < 32; ++e) {
    qp_h[o32 + e] = (_Float16)0.f;
    kp_h[o32 + e] = (_Float16)0.f;
  }
#pragma unroll
  for (int e = 0; e < 12; ++e)
    vpT[((size_t)bh * 16 + e) * kN + n] = (_Float16)vpv[e];
#pragma unroll
  for (int e = 12; e < 16; ++e)
    vpT[((size_t)bh * 16 + e) * kN + n] = (_Float16)0.f;
  q2[(size_t)bh * kN + n] = q2s;
  k2[(size_t)bh * kN + n] = k2s;
}

// ---------------------------------------------------------------------------
// Kernel: fused flash-style IPA attention. One wave = one (b,h, 16-row tile).
// Per 32-key step: 6 WMMAs for logits (QK d=64 + point d=12pad32),
// online softmax with ds_bpermute row reductions, P->f16 via per-wave LDS
// tile, then 5 WMMAs for attn x [vs | vp]. Logits tensor (64MB) never
// touches HBM.
// ---------------------------------------------------------------------------
__global__ void __launch_bounds__(256)
ipa_attention(const _Float16* __restrict__ qs_h, const _Float16* __restrict__ ks_h,
              const _Float16* __restrict__ vsT, const _Float16* __restrict__ qp_h,
              const _Float16* __restrict__ kp_h, const _Float16* __restrict__ vpT,
              const float* __restrict__ q2, const float* __restrict__ k2,
              const float* __restrict__ pwc, float* __restrict__ rs,
              float* __restrict__ rp) {
  __shared__ _Float16 sP[8][16 * 32];   // per-wave 16x32 probability tile

  const int lane = threadIdx.x & 31;
  const int w    = threadIdx.x >> 5;
  const int wid  = blockIdx.x * 8 + w;
  const int it   = wid & 63;
  const int bh   = wid >> 6;
  const int h    = bh % kH;
  const int i0   = it * 16;
  const int g    = lane >> 4;

  const float sscale = rsqrtf(2.f * kDH);
  const float pw2    = pwc[h];

  const _Float16* qb  = qs_h + ((size_t)bh * kN + i0) * kDH;
  const _Float16* qpb = qp_h + ((size_t)bh * kN + i0) * 32;
  const _Float16* kb  = ks_h + (size_t)bh * kN * kDH;
  const _Float16* kpb = kp_h + (size_t)bh * kN * 32;
  const _Float16* vb  = vsT + (size_t)bh * kDH * kN;
  const _Float16* vpb = vpT + (size_t)bh * 16 * kN;

  const v16h aq0 = load_a_frag(qb, kDH, 0, lane);
  const v16h aq1 = load_a_frag(qb, kDH, 32, lane);
  const v16h aqp = load_a_frag(qpb, 32, 0, lane);
  const v8f  q2v = *reinterpret_cast<const v8f*>(q2 + (size_t)bh * kN + i0 + 8 * g);

  v8f accs[4] = {};
  v8f accp    = {};
  float mrun[8], lrun[8];
#pragma unroll
  for (int r = 0; r < 8; ++r) { mrun[r] = -1e30f; lrun[r] = 0.f; }

  for (int j0 = 0; j0 < kN; j0 += 32) {
    if (j0 + 32 < kN) {   // pull next key/value tiles toward L2/L0
      const int jn = j0 + 32 + (lane & 15);
      __builtin_prefetch(kb + (size_t)jn * kDH, 0, 3);
      __builtin_prefetch(kpb + (size_t)jn * 32, 0, 3);
      __builtin_prefetch(vb + (size_t)(lane & 15) * kN + j0 + 32, 0, 3);
    }
    v8f lg[2];
#pragma unroll
    for (int t = 0; t < 2; ++t) {
      const int jt = j0 + 16 * t;
      v16h bs0 = load_b_frag(kb, kDH, jt, 0, lane);
      v16h bs1 = load_b_frag(kb, kDH, jt, 32, lane);
      v8f  cs  = {};
      cs = wmma16(aq0, bs0, cs);
      cs = wmma16(aq1, bs1, cs);
      v16h bp = load_b_frag(kpb, 32, jt, 0, lane);
      v8f  cq = {};
      cq = wmma16(aqp, bp, cq);
      const float k2t = k2[(size_t)bh * kN + jt + (lane & 15)];
#pragma unroll
      for (int r = 0; r < 8; ++r)
        lg[t][r] = cs[r] * sscale - pw2 * (q2v[r] + k2t - 2.f * cq[r]);
    }

    // online softmax (rows live across 16-lane groups)
    float p0[8], p1[8], al[8];
#pragma unroll
    for (int r = 0; r < 8; ++r) {
      float mt = redmax16(fmaxf(lg[0][r], lg[1][r]), lane);
      float nm = fmaxf(mrun[r], mt);
      al[r]    = __expf(mrun[r] - nm);
      p0[r]    = __expf(lg[0][r] - nm);
      p1[r]    = __expf(lg[1][r] - nm);
      float s  = redsum16(p0[r] + p1[r], lane);
      lrun[r]  = lrun[r] * al[r] + s;
      mrun[r]  = nm;
    }
#pragma unroll
    for (int r = 0; r < 8; ++r) {
      accs[0][r] *= al[r]; accs[1][r] *= al[r];
      accs[2][r] *= al[r]; accs[3][r] *= al[r];
      accp[r]    *= al[r];
    }

    // P (f32, C layout) -> LDS 16x32 f16 tile -> A fragment
    {
      _Float16* sp  = sP[w];
      const int col = lane & 15;
#pragma unroll
      for (int r = 0; r < 8; ++r) {
        const int row = r + 8 * g;
        sp[row * 32 + col]      = (_Float16)p0[r];
        sp[row * 32 + 16 + col] = (_Float16)p1[r];
      }
    }
    v16h pa = load_a_frag(sP[w], 32, 0, lane);   // same-wave LDS RAW, in-order

#pragma unroll
    for (int dc = 0; dc < 4; ++dc) {
      v16h bv  = load_b_frag(vb, kN, dc * 16, j0, lane);
      accs[dc] = wmma16(pa, bv, accs[dc]);
    }
    v16h bvp = load_b_frag(vpb, kN, 0, j0, lane);
    accp = wmma16(pa, bvp, accp);
  }

  // normalize + store
  float inv[8];
#pragma unroll
  for (int r = 0; r < 8; ++r) inv[r] = 1.f / lrun[r];
  const int    col  = lane & 15;
  const size_t rowb = (size_t)bh * kN + i0 + 8 * g;
#pragma unroll
  for (int r = 0; r < 8; ++r) {
    const size_t row = rowb + r;
#pragma unroll
    for (int dc = 0; dc < 4; ++dc)
      rs[row * kDH + dc * 16 + col] = accs[dc][r] * inv[r];
    rp[row * 16 + col] = accp[r] * inv[r];
  }
}

// ---------------------------------------------------------------------------
// Kernel: inverse rotation, point norms, concat -> f16 (2048 x 640)
// ---------------------------------------------------------------------------
__global__ void finish_concat(const float* __restrict__ rs,
                              const float* __restrict__ rp,
                              const float* __restrict__ rot,
                              const float* __restrict__ trans,
                              _Float16* __restrict__ cat) {
  int m = blockIdx.x * blockDim.x + threadIdx.x;
  if (m >= kM) return;
  int b = m / kN, n = m % kN;
  const float* R = rot + (size_t)m * 9;
  const float* t = trans + (size_t)m * 3;
  _Float16* crow = cat + (size_t)m * kCat;

  for (int h = 0; h < kH; ++h) {
    int bh = b * kH + h;
    const float* rsrow = rs + ((size_t)bh * kN + n) * kDH;
#pragma unroll 4
    for (int d = 0; d < kDH; ++d) crow[h * kDH + d] = (_Float16)rsrow[d];

    const float* rprow = rp + ((size_t)bh * kN + n) * 16;
#pragma unroll
    for (int d = 0; d < kPD; ++d) {
      float pin[3];
#pragma unroll
      for (int r = 0; r < 3; ++r) pin[r] = rprow[d * 3 + r] - t[r];
      float s2 = kEPS;
      float out3[3];
#pragma unroll
      for (int c = 0; c < 3; ++c) {
        float v = 0.f;
#pragma unroll
        for (int r = 0; r < 3; ++r) v += R[r * 3 + c] * pin[r];  // R^T
        out3[c] = v;
        s2 += v * v;
      }
#pragma unroll
      for (int c = 0; c < 3; ++c)
        crow[kSDim + h * kPD * 3 + d * 3 + c] = (_Float16)out3[c];
      crow[kSDim + kPDim + h * kPD + d] = (_Float16)sqrtf(s2);
    }
  }
}

// ---------------------------------------------------------------------------
extern "C" void kernel_launch(void* const* d_in, const int* in_sizes, int n_in,
                              void* d_out, int out_size, void* d_ws,
                              size_t ws_size, hipStream_t stream) {
  const float* x     = (const float*)d_in[0];
  const float* rot   = (const float*)d_in[1];
  const float* trans = (const float*)d_in[2];
  const float* Wsq   = (const float*)d_in[3];
  const float* Wsk   = (const float*)d_in[4];
  const float* Wsv   = (const float*)d_in[5];
  const float* Wpq   = (const float*)d_in[6];
  const float* Wpk   = (const float*)d_in[7];
  const float* Wpv   = (const float*)d_in[8];
  const float* pw    = (const float*)d_in[9];
  const float* Wout  = (const float*)d_in[10];
  const float* bout  = (const float*)d_in[11];
  float*       out   = (float*)d_out;

  char*  ws  = (char*)d_ws;
  size_t off = 0;
  auto alloc = [&](size_t bytes) -> char* {
    char* p = ws + off;
    off = (off + bytes + 255) & ~(size_t)255;
    return p;
  };

  const int BHN = kB * kH * kN;
  _Float16* xh    = (_Float16*)alloc((size_t)kM * kDIM * 2);
  _Float16* wsqh  = (_Float16*)alloc((size_t)kSDim * kDIM * 2);
  _Float16* wskh  = (_Float16*)alloc((size_t)kSDim * kDIM * 2);
  _Float16* wsvh  = (_Float16*)alloc((size_t)kSDim * kDIM * 2);
  _Float16* wpqh  = (_Float16*)alloc((size_t)kPDim * kDIM * 2);
  _Float16* wpkh  = (_Float16*)alloc((size_t)kPDim * kDIM * 2);
  _Float16* wpvh  = (_Float16*)alloc((size_t)kPDim * kDIM * 2);
  _Float16* wouth = (_Float16*)alloc((size_t)kDIM * kCat * 2);
  float* Psq = (float*)alloc((size_t)kM * kSDim * 4);
  float* Psk = (float*)alloc((size_t)kM * kSDim * 4);
  float* Psv = (float*)alloc((size_t)kM * kSDim * 4);
  float* Ppq = (float*)alloc((size_t)kM * kPDim * 4);
  float* Ppk = (float*)alloc((size_t)kM * kPDim * 4);
  float* Ppv = (float*)alloc((size_t)kM * kPDim * 4);
  _Float16* qs_h = (_Float16*)alloc((size_t)BHN * kDH * 2);
  _Float16* ks_h = (_Float16*)alloc((size_t)BHN * kDH * 2);
  _Float16* vsT  = (_Float16*)alloc((size_t)BHN * kDH * 2);
  _Float16* qp_h = (_Float16*)alloc((size_t)BHN * 32 * 2);
  _Float16* kp_h = (_Float16*)alloc((size_t)BHN * 32 * 2);
  _Float16* vpT  = (_Float16*)alloc((size_t)BHN * 16 * 2);
  float* q2  = (float*)alloc((size_t)BHN * 4);
  float* k2  = (float*)alloc((size_t)BHN * 4);
  float* pwc = (float*)alloc(kH * 4);
  float* rsb = (float*)alloc((size_t)BHN * kDH * 4);
  float* rpb = (float*)alloc((size_t)BHN * 16 * 4);
  _Float16* cat = (_Float16*)alloc((size_t)kM * kCat * 2);

  auto cvt = [&](const float* in, _Float16* o, int n) {
    int blocks = (n + 255) / 256;
    if (blocks > 2048) blocks = 2048;
    f32_to_f16<<<blocks, 256, 0, stream>>>(in, o, n);
  };
  cvt(x, xh, kM * kDIM);
  cvt(Wsq, wsqh, kSDim * kDIM);
  cvt(Wsk, wskh, kSDim * kDIM);
  cvt(Wsv, wsvh, kSDim * kDIM);
  cvt(Wpq, wpqh, kPDim * kDIM);
  cvt(Wpk, wpkh, kPDim * kDIM);
  cvt(Wpv, wpvh, kPDim * kDIM);
  cvt(Wout, wouth, kDIM * kCat);
  softplus_pw<<<1, 32, 0, stream>>>(pw, pwc);

  // 32x64-per-wave register-blocked GEMMs (grids exact: 8 waves/block)
  auto gemm64 = [&](const _Float16* A, const _Float16* Bw, const float* bias,
                    float* C, int K, int Nout) {
    int waves = (kM / 32) * (Nout / 64);
    gemm_f16f32<2, 4><<<waves / 8, 256, 0, stream>>>(A, Bw, bias, C, K, Nout);
  };
  auto gemm48 = [&](const _Float16* A, const _Float16* Bw, const float* bias,
                    float* C, int K, int Nout) {
    int waves = (kM / 32) * (Nout / 48);
    gemm_f16f32<2, 3><<<waves / 8, 256, 0, stream>>>(A, Bw, bias, C, K, Nout);
  };
  gemm64(xh, wsqh, nullptr, Psq, kDIM, kSDim);
  gemm64(xh, wskh, nullptr, Psk, kDIM, kSDim);
  gemm64(xh, wsvh, nullptr, Psv, kDIM, kSDim);
  gemm48(xh, wpqh, nullptr, Ppq, kDIM, kPDim);
  gemm48(xh, wpkh, nullptr, Ppk, kDIM, kPDim);
  gemm48(xh, wpvh, nullptr, Ppv, kDIM, kPDim);

  reshape_rotate<<<(kB * kN * kH + 255) / 256, 256, 0, stream>>>(
      Psq, Psk, Psv, Ppq, Ppk, Ppv, rot, trans, qs_h, ks_h, vsT, qp_h, kp_h,
      vpT, q2, k2);

  ipa_attention<<<(kB * kH * (kN / 16)) / 8, 256, 0, stream>>>(
      qs_h, ks_h, vsT, qp_h, kp_h, vpT, q2, k2, pwc, rsb, rpb);

  finish_concat<<<(kM + 255) / 256, 256, 0, stream>>>(rsb, rpb, rot, trans, cat);

  gemm64(cat, wouth, bout, out, kCat, kDIM);
}
